// GCN2_4784593568268
// MI455X (gfx1250) — matrix operand order repaired and checked by reading, not compiled, not run on hardware
//
#include <hip/hip_runtime.h>

// ---------------------------------------------------------------------------
// GCN 2-layer forward for MI455X (gfx1250, wave32).
//  - GEMMs via v_wmma_f32_16x16x32_f16 with hi/lo f16 error-compensated split
//    (fp32-grade accuracy, matrix-core throughput).
//  - Edge aggregation via global float atomics; all tensors fit in 192MB L2.
// ---------------------------------------------------------------------------

typedef _Float16 f16;
typedef __attribute__((ext_vector_type(16))) _Float16 v16h;
typedef __attribute__((ext_vector_type(8)))  float    v8f;

union Frag { v16h v; uint4 u[2]; };

__device__ __forceinline__ void splitf(float v, f16* hi, f16* lo) {
  f16 h = (f16)v;            // RTNE f32->f16
  *hi = h;
  *lo = (f16)(v - (float)h); // residual
}

// ---------------- elementwise / prep kernels -------------------------------

__global__ void k_fill_f32(float* __restrict__ p, float v, int n) {
  int i = blockIdx.x * blockDim.x + threadIdx.x;
  if (i < n) p[i] = v;
}

__global__ void k_degree(const long long* __restrict__ dst, float* __restrict__ deg, int E) {
  int i = blockIdx.x * blockDim.x + threadIdx.x;
  int stride = gridDim.x * blockDim.x;
  for (; i < E; i += stride) atomicAdd(&deg[(int)dst[i]], 1.0f);
}

__global__ void k_rsqrt(float* __restrict__ p, int n) {
  int i = blockIdx.x * blockDim.x + threadIdx.x;
  if (i < n) p[i] = rsqrtf(p[i]);
}

__global__ void k_split(const float* __restrict__ in, f16* __restrict__ hi,
                        f16* __restrict__ lo, int n) {
  int i = blockIdx.x * blockDim.x + threadIdx.x;
  if (i < n) splitf(in[i], &hi[i], &lo[i]);
}

// W[k*Nout + n] (row-major K x Nout) -> Wt hi/lo at [n*K + k] (Nout x K)
__global__ void k_split_transpose(const float* __restrict__ W, f16* __restrict__ hi,
                                  f16* __restrict__ lo, int K, int Nout) {
  int i = blockIdx.x * blockDim.x + threadIdx.x;
  if (i < K * Nout) {
    int k = i / Nout, n = i - k * Nout;
    splitf(W[i], &hi[(size_t)n * K + k], &lo[(size_t)n * K + k]);
  }
}

__global__ void k_relu_bias_split(const float* __restrict__ acc, const float* __restrict__ b,
                                  f16* __restrict__ hi, f16* __restrict__ lo, int n, int F) {
  int i = blockIdx.x * blockDim.x + threadIdx.x;
  if (i < n) {
    float v = acc[i] + b[i % F];
    v = v > 0.f ? v : 0.f;
    splitf(v, &hi[i], &lo[i]);
  }
}

__global__ void k_bias_out(const float* __restrict__ acc, const float* __restrict__ b,
                           float* __restrict__ out, int n, int F) {
  int i = blockIdx.x * blockDim.x + threadIdx.x;
  if (i < n) out[i] = acc[i] + b[i % F];
}

// ---------------- WMMA GEMM: H = A @ B^T, Acc = H * dinv[row]^2 ------------
// A (hi/lo): [M x K] f16 row-major.  B (hi/lo): [Nout x K] f16 row-major
// (i.e. W transposed).  One wave computes one 16x16 tile of H.
// A-frag (ISA 16-bit A 16x32): lane<16 -> K {0..7,16..23}, lane>=16 -> +8.
// B-frag (ISA 16-bit B 32x16): lane<16 -> K {0..7,16..23}, lane>=16 -> +8,
//   column = lane&15; same K order as A-frag.
// C/D: VGPR r -> row = r + 8*(lane>=16), col = lane&15.

__global__ void __launch_bounds__(256)
k_gemm_wmma(const f16* __restrict__ Ahi, const f16* __restrict__ Alo,
            const f16* __restrict__ Bhi, const f16* __restrict__ Blo,
            float* __restrict__ H, float* __restrict__ Acc,
            const float* __restrict__ dinv, int M, int K, int Nout)
{
  const int lane = threadIdx.x & 31;
  const int wave = (blockIdx.x << 3) + (threadIdx.x >> 5);
  const int tilesN = Nout >> 4;
  const int tiles  = (M >> 4) * tilesN;
  if (wave >= tiles) return;                 // wave-uniform: EXEC stays all-1s
  const int tM = wave / tilesN;
  const int tN = wave - tM * tilesN;

  const int mrow = (tM << 4) + (lane & 15);
  const int ncol = (tN << 4) + (lane & 15);
  const int ka = (lane >= 16) ? 8 : 0;
  const int kb = (lane >= 16) ? 8 : 0;

  const f16* arh = Ahi + (size_t)mrow * K;
  const f16* arl = Alo + (size_t)mrow * K;
  const f16* brh = Bhi + (size_t)ncol * K;
  const f16* brl = Blo + (size_t)ncol * K;

  v8f acc = {};
  for (int k0 = 0; k0 < K; k0 += 32) {
    Frag ah, al, bh, bl;
    ah.u[0] = *(const uint4*)(arh + k0 + ka);
    ah.u[1] = *(const uint4*)(arh + k0 + ka + 16);
    al.u[0] = *(const uint4*)(arl + k0 + ka);
    al.u[1] = *(const uint4*)(arl + k0 + ka + 16);
    bh.u[0] = *(const uint4*)(brh + k0 + kb);
    bh.u[1] = *(const uint4*)(brh + k0 + kb + 16);
    bl.u[0] = *(const uint4*)(brl + k0 + kb);
    bl.u[1] = *(const uint4*)(brl + k0 + kb + 16);
    // hi*hi + hi*lo + lo*hi  (error-compensated f16 -> ~fp32 accuracy)
    acc = __builtin_amdgcn_wmma_f32_16x16x32_f16(false, ah.v, false, bh.v, (short)0, acc, false, false);
    acc = __builtin_amdgcn_wmma_f32_16x16x32_f16(false, ah.v, false, bl.v, (short)0, acc, false, false);
    acc = __builtin_amdgcn_wmma_f32_16x16x32_f16(false, al.v, false, bh.v, (short)0, acc, false, false);
  }

  const int baseRow = (tM << 4) + ((lane >= 16) ? 8 : 0);
#pragma unroll
  for (int r = 0; r < 8; ++r) {
    const int row = baseRow + r;
    const float v  = acc[r];
    const float di = dinv[row];
    H  [(size_t)row * Nout + ncol] = v;          // raw h (message source)
    Acc[(size_t)row * Nout + ncol] = v * di * di; // self-loop init of aggregator
  }
}

// ---------------- edge scatter: Acc[dst] += dinv[s]*dinv[d]*H[src] ---------
template<int FPL> // floats handled per lane (F/32)
__global__ void __launch_bounds__(256)
k_edge_scatter(const long long* __restrict__ src, const long long* __restrict__ dst,
               const float* __restrict__ dinv, const float* __restrict__ H,
               float* __restrict__ Acc, int E, int F)
{
  const int lane = threadIdx.x & 31;
  int wave = (blockIdx.x << 3) + (threadIdx.x >> 5);
  const int nWaves = gridDim.x << 3;
  for (; wave < E; wave += nWaves) {
    const int s = (int)src[wave];
    const int d = (int)dst[wave];
    const float coef = dinv[s] * dinv[d];
    const float* hs = H + (size_t)s * F + lane * FPL;
    float* ad = Acc + (size_t)d * F + lane * FPL;
    if constexpr (FPL == 4) {
      const float4 v = *(const float4*)hs;
      atomicAdd(ad + 0, coef * v.x);
      atomicAdd(ad + 1, coef * v.y);
      atomicAdd(ad + 2, coef * v.z);
      atomicAdd(ad + 3, coef * v.w);
    } else {
      const float2 v = *(const float2*)hs;
      atomicAdd(ad + 0, coef * v.x);
      atomicAdd(ad + 1, coef * v.y);
    }
  }
}

// ---------------------------------------------------------------------------

extern "C" void kernel_launch(void* const* d_in, const int* in_sizes, int n_in,
                              void* d_out, int out_size, void* d_ws, size_t ws_size,
                              hipStream_t stream) {
  const float*     x  = (const float*)d_in[0];
  const long long* ei = (const long long*)d_in[1];  // int64 edge_index [2,E]
  const float*     W1 = (const float*)d_in[2];
  const float*     b1 = (const float*)d_in[3];
  const float*     W2 = (const float*)d_in[4];
  const float*     b2 = (const float*)d_in[5];

  const int D_IN = 128, D_HID = 128, D_OUT = 64;
  const int N = in_sizes[0] / D_IN;   // 50000 (divisible by 16)
  const int E = in_sizes[1] / 2;      // 800000
  const long long* src = ei;
  const long long* dst = ei + E;

  char* base = (char*)d_ws;
  size_t off = 0;
  auto alloc = [&](size_t bytes) -> void* {
    void* p = base + off;
    off = (off + bytes + 255) & ~(size_t)255;
    return p;
  };

  float* dinv = (float*)alloc((size_t)N * 4);                 // deg -> dinv in place
  f16*   xhi  = (f16*)alloc((size_t)N * D_IN * 2);
  f16*   xlo  = (f16*)alloc((size_t)N * D_IN * 2);
  f16*   w1hi = (f16*)alloc((size_t)D_IN * D_HID * 2);
  f16*   w1lo = (f16*)alloc((size_t)D_IN * D_HID * 2);
  f16*   w2hi = (f16*)alloc((size_t)D_HID * D_OUT * 2);
  f16*   w2lo = (f16*)alloc((size_t)D_HID * D_OUT * 2);
  float* h1   = (float*)alloc((size_t)N * D_HID * 4);
  float* acc1 = (float*)alloc((size_t)N * D_HID * 4);
  f16*   h1hi = (f16*)alloc((size_t)N * D_HID * 2);
  f16*   h1lo = (f16*)alloc((size_t)N * D_HID * 2);
  float* h2   = (float*)alloc((size_t)N * D_OUT * 4);
  float* acc2 = (float*)alloc((size_t)N * D_OUT * 4);
  (void)ws_size; (void)n_in; (void)out_size;

  auto blocks = [](long long n) { return (unsigned)((n + 255) / 256); };

  // degrees (self-loop counted via init to 1.0), then dinv = rsqrt(deg)
  k_fill_f32<<<blocks(N), 256, 0, stream>>>(dinv, 1.0f, N);
  k_degree<<<2048, 256, 0, stream>>>(dst, dinv, E);
  k_rsqrt<<<blocks(N), 256, 0, stream>>>(dinv, N);

  // fp32 -> f16 hi/lo operands (weights transposed to [Nout x K])
  k_split<<<blocks((long long)N * D_IN), 256, 0, stream>>>(x, xhi, xlo, N * D_IN);
  k_split_transpose<<<blocks(D_IN * D_HID), 256, 0, stream>>>(W1, w1hi, w1lo, D_IN, D_HID);
  k_split_transpose<<<blocks(D_HID * D_OUT), 256, 0, stream>>>(W2, w2hi, w2lo, D_HID, D_OUT);

  // ---- layer 1 ----
  const int tiles1 = (N / 16) * (D_HID / 16);
  k_gemm_wmma<<<(tiles1 + 7) / 8, 256, 0, stream>>>(xhi, xlo, w1hi, w1lo,
                                                    h1, acc1, dinv, N, D_IN, D_HID);
  k_edge_scatter<4><<<4096, 256, 0, stream>>>(src, dst, dinv, h1, acc1, E, D_HID);
  k_relu_bias_split<<<blocks((long long)N * D_HID), 256, 0, stream>>>(acc1, b1, h1hi, h1lo,
                                                                      N * D_HID, D_HID);

  // ---- layer 2 ----
  const int tiles2 = (N / 16) * (D_OUT / 16);
  k_gemm_wmma<<<(tiles2 + 7) / 8, 256, 0, stream>>>(h1hi, h1lo, w2hi, w2lo,
                                                    h2, acc2, dinv, N, D_HID, D_OUT);
  k_edge_scatter<2><<<4096, 256, 0, stream>>>(src, dst, dinv, h2, acc2, E, D_OUT);
  k_bias_out<<<blocks((long long)N * D_OUT), 256, 0, stream>>>(acc2, b2, (float*)d_out,
                                                               N * D_OUT, D_OUT);
}
